// CustomResidualAttentionBlock_83305185673357
// MI455X (gfx1250) — compile-verified, matrix operand order checked by reading, not run on hardware
//
#include <hip/hip_runtime.h>
#include <hip/hip_bf16.h>
#include <stdint.h>

// CustomResidualAttentionBlock (open_clip scaled-cosine attention) for MI455X / gfx1250.
// All GEMMs + attention matmuls run on v_wmma_f32_16x16x32_bf16 (wave32).
// f32 is kept for LayerNorm / l2-norm / softmax / accumulators.
// Attention softmax uses a fixed per-head shift (valid because |S| <= logit_scale,
// softmax is shift-invariant) so the streaming loop has NO cross-lane reductions.

#define B_   2
#define L_   2048
#define C_   1024
#define H_   16
#define HD_  64
#define MLP_ 4096
#define LOGMAX 4.6051701859880914f   // log(100)
#define LN_EPS 1e-5f

typedef __attribute__((ext_vector_type(16))) __bf16 v16bf;
typedef __attribute__((ext_vector_type(8)))  float  v8f;

union Frag {
  v16bf v;
  uint4 q[2];
};

static __device__ __forceinline__ v8f wmma_bf16(v16bf a, v16bf b, v8f c) {
  // D = A(16x32 bf16) x B(32x16 bf16) + C(16x16 f32)
  return __builtin_amdgcn_wmma_f32_16x16x32_bf16(false, a, false, b, (short)0, c,
                                                 false, false);
}

static __device__ __forceinline__ unsigned short f2bf(float f) {
  unsigned int u = __float_as_uint(f);
  u += 0x7FFFu + ((u >> 16) & 1u);   // round-to-nearest-even
  return (unsigned short)(u >> 16);
}

// ---------------------------------------------------------------- cvt f32->bf16
__global__ void k_cvt_bf16(const float* __restrict__ src,
                           unsigned short* __restrict__ dst, int n) {
  int i = blockIdx.x * 256 + threadIdx.x;
  if (i < n) dst[i] = f2bf(src[i]);
}

// ---------------------------------------------------------------- LayerNorm -> bf16
// One block (256 threads) per row of length C_=1024.
__global__ void k_ln_bf16(const float* __restrict__ x, const float* __restrict__ g,
                          const float* __restrict__ bt, unsigned short* __restrict__ out) {
  __shared__ float red[16];
  const int tid = threadIdx.x;
  const int lane = tid & 31, wave = tid >> 5;
  const size_t row = blockIdx.x;
  const float* xr = x + row * C_;
  float vals[4], s = 0.f, sq = 0.f;
#pragma unroll
  for (int j = 0; j < 4; ++j) {
    float v = xr[tid + j * 256];
    vals[j] = v; s += v; sq += v * v;
  }
#pragma unroll
  for (int off = 16; off > 0; off >>= 1) { s += __shfl_xor(s, off); sq += __shfl_xor(sq, off); }
  if (lane == 0) { red[wave] = s; red[8 + wave] = sq; }
  __syncthreads();
  float st = 0.f, sqt = 0.f;
#pragma unroll
  for (int w = 0; w < 8; ++w) { st += red[w]; sqt += red[8 + w]; }
  const float mu = st * (1.f / C_);
  const float var = sqt * (1.f / C_) - mu * mu;
  const float rstd = rsqrtf(var + LN_EPS);
#pragma unroll
  for (int j = 0; j < 4; ++j) {
    int i = tid + j * 256;
    out[row * C_ + i] = f2bf((vals[j] - mu) * rstd * g[i] + bt[i]);
  }
}

// ---------------------------------------------------------------- x2 = x + ls1 * LN(a)
__global__ void k_resid_ln(const float* __restrict__ a, const float* __restrict__ x,
                           const float* __restrict__ g, const float* __restrict__ bt,
                           const float* __restrict__ ls, float* __restrict__ x2) {
  __shared__ float red[16];
  const int tid = threadIdx.x;
  const int lane = tid & 31, wave = tid >> 5;
  const size_t row = blockIdx.x;
  const float* ar = a + row * C_;
  float vals[4], s = 0.f, sq = 0.f;
#pragma unroll
  for (int j = 0; j < 4; ++j) {
    float v = ar[tid + j * 256];
    vals[j] = v; s += v; sq += v * v;
  }
#pragma unroll
  for (int off = 16; off > 0; off >>= 1) { s += __shfl_xor(s, off); sq += __shfl_xor(sq, off); }
  if (lane == 0) { red[wave] = s; red[8 + wave] = sq; }
  __syncthreads();
  float st = 0.f, sqt = 0.f;
#pragma unroll
  for (int w = 0; w < 8; ++w) { st += red[w]; sqt += red[8 + w]; }
  const float mu = st * (1.f / C_);
  const float rstd = rsqrtf(sqt * (1.f / C_) - mu * mu + LN_EPS);
#pragma unroll
  for (int j = 0; j < 4; ++j) {
    int i = tid + j * 256;
    float ln = (vals[j] - mu) * rstd * g[i] + bt[i];
    x2[row * C_ + i] = x[row * C_ + i] + ls[i] * ln;
  }
}

// ---------------------------------------------------------------- WMMA GEMM
// C[M,N] = epi( A_bf16[M,K] @ W_bf16[N,K]^T + bias[N] )
// Block 256 = 8 waves; wave w -> rows [by*128 + w*16, +16), cols [bx*64, +64).
// The 64x32 W strip for each k-step is staged in LDS once per block (8x less
// global B traffic than per-wave loads).
// EPI: 0 = f32 store, 1 = bf16 store with exact GELU, 2 = f32 store out = resid + ls*val.
template <int EPI>
__global__ void k_gemm(const unsigned short* __restrict__ A,
                       const unsigned short* __restrict__ W,
                       const float* __restrict__ bias, int M, int N, int K,
                       float* __restrict__ outf, unsigned short* __restrict__ outbf,
                       const float* __restrict__ resid, const float* __restrict__ lsv) {
  __shared__ alignas(16) unsigned short wbuf[64 * 32];   // 4 KB
  const int tid = threadIdx.x;
  const int lane = tid & 31, wave = tid >> 5;
  const int hi = lane >> 4, ln16 = lane & 15;
  const int m0 = blockIdx.y * 128 + wave * 16;
  const int n0 = blockIdx.x * 64;
  // staging role: thread t loads 8 bf16 of W row (n0 + t/4), k-chunk (t%4)*8
  const int sr = tid >> 2, sc = (tid & 3) * 8;
  const unsigned short* wp = W + (size_t)(n0 + sr) * K + sc;
  v8f acc[4] = {};
  // A fragment: lane holds row (m0+ln16); half 'hi' supplies K-chunks {hi*8..+7, 16+hi*8..+7}
  const unsigned short* ap = A + (size_t)(m0 + ln16) * K + hi * 8;
  for (int k0 = 0; k0 < K; k0 += 32) {
    __syncthreads();                                      // protect previous reads
    *reinterpret_cast<uint4*>(&wbuf[sr * 32 + sc]) =
        *reinterpret_cast<const uint4*>(wp + k0);
    __syncthreads();
    Frag fa;
    fa.q[0] = *reinterpret_cast<const uint4*>(ap + k0);
    fa.q[1] = *reinterpret_cast<const uint4*>(ap + k0 + 16);
#pragma unroll
    for (int nn = 0; nn < 4; ++nn) {
      const unsigned short* bp = &wbuf[(nn * 16 + ln16) * 32 + hi * 8];
      Frag fb;
      fb.q[0] = *reinterpret_cast<const uint4*>(bp);
      fb.q[1] = *reinterpret_cast<const uint4*>(bp + 16);
      acc[nn] = wmma_bf16(fa.v, fb.v, acc[nn]);
    }
  }
#pragma unroll
  for (int nn = 0; nn < 4; ++nn) {
    const int n = n0 + nn * 16 + ln16;
    const float bz = bias[n];
#pragma unroll
    for (int v = 0; v < 8; ++v) {
      const int m = m0 + v + 8 * hi;           // C-layout: VGPR v -> row v (+8 for hi half)
      float val = acc[nn][v] + bz;
      size_t idx = (size_t)m * N + n;
      if (EPI == 0) {
        outf[idx] = val;
      } else if (EPI == 1) {
        outbf[idx] = f2bf(0.5f * val * (1.f + erff(val * 0.70710678118f)));
      } else {
        outf[idx] = resid[idx] + lsv[n] * val;
      }
    }
  }
}

// ---------------------------------------------------------------- QKV split + l2-norm
// qn/kn: [B,H,L,HD] bf16 (q scaled by exp(min(logit_scale,log100))); vT: [B,H,HD,L] bf16.
__global__ void k_split_qkv(const float* __restrict__ qkv,
                            const float* __restrict__ logit_scale,
                            unsigned short* __restrict__ qn,
                            unsigned short* __restrict__ kn,
                            unsigned short* __restrict__ vT) {
  const int lane = threadIdx.x & 31, wave = threadIdx.x >> 5;
  const int row = blockIdx.x * 8 + wave;       // (b*H + h)*L + l
  const int l = row & (L_ - 1);
  const int bh = row >> 11;                    // L_ = 2048
  const int h = bh & (H_ - 1);
  const int b = bh >> 4;
  const float* base = qkv + ((size_t)(b * L_ + l)) * (3 * C_) + h * HD_;
  const int d0 = 2 * lane, d1 = 2 * lane + 1;
  float q0 = base[d0],          q1 = base[d1];
  float k0 = base[C_ + d0],     k1 = base[C_ + d1];
  float v0 = base[2 * C_ + d0], v1 = base[2 * C_ + d1];
  float sqq = q0 * q0 + q1 * q1, sqk = k0 * k0 + k1 * k1;
#pragma unroll
  for (int off = 16; off > 0; off >>= 1) {
    sqq += __shfl_xor(sqq, off);
    sqk += __shfl_xor(sqk, off);
  }
  const float ls = __expf(fminf(logit_scale[h], LOGMAX));
  const float iq = ls / fmaxf(sqrtf(sqq), 1e-12f);
  const float ik = 1.f / fmaxf(sqrtf(sqk), 1e-12f);
  const size_t ro = (size_t)row * HD_;
  qn[ro + d0] = f2bf(q0 * iq); qn[ro + d1] = f2bf(q1 * iq);
  kn[ro + d0] = f2bf(k0 * ik); kn[ro + d1] = f2bf(k1 * ik);
  const size_t vo = (size_t)bh * HD_ * L_;
  vT[vo + (size_t)d0 * L_ + l] = f2bf(v0);
  vT[vo + (size_t)d1 * L_ + l] = f2bf(v1);
}

// ---------------------------------------------------------------- flash attention
// One wave per 16-query tile of one (b,h). S via WMMA (K=64 split in two 16x16x32),
// fixed-shift streaming softmax (|S| <= ls so shift = min(ls,30) is safe and the
// normalized result equals the max-shifted softmax), P re-shaped via LDS to an
// A-fragment, O accumulated with 4 WMMAs per 32 keys. head_scale folded into writeout.
__global__ void k_flash(const unsigned short* __restrict__ qn,
                        const unsigned short* __restrict__ kn,
                        const unsigned short* __restrict__ vT,
                        const float* __restrict__ logit_scale,
                        const float* __restrict__ head_scale,
                        unsigned short* __restrict__ O) {
  __shared__ alignas(16) unsigned short pbuf[8][16 * 32];
  const int lane = threadIdx.x & 31, wave = threadIdx.x >> 5;
  const int hi = lane >> 4, ln16 = lane & 15;
  const int bh = blockIdx.y;
  const int h = bh & (H_ - 1);
  const int b = bh >> 4;
  const int qtile = blockIdx.x * 8 + wave;
  const unsigned short* qp = qn + ((size_t)bh * L_ + qtile * 16 + ln16) * HD_ + hi * 8;
  Frag qa0, qa1;                       // q rows, hd 0..31 / 32..63
  qa0.q[0] = *(const uint4*)(qp);      qa0.q[1] = *(const uint4*)(qp + 16);
  qa1.q[0] = *(const uint4*)(qp + 32); qa1.q[1] = *(const uint4*)(qp + 48);
  const unsigned short* kbase = kn + (size_t)bh * L_ * HD_;
  const unsigned short* vbase = vT + (size_t)bh * HD_ * L_;
  const float shift = fminf(__expf(fminf(logit_scale[h], LOGMAX)), 30.f);
  v8f oacc[4] = {};
  float lst[8];
#pragma unroll
  for (int v = 0; v < 8; ++v) lst[v] = 0.f;
  unsigned short* prow = &pbuf[wave][0];

  for (int j = 0; j < L_; j += 32) {
    v8f s0 = {}, s1 = {};
    {
      const unsigned short* kp = kbase + (size_t)(j + ln16) * HD_ + hi * 8;
      Frag b0, b1;
      b0.q[0] = *(const uint4*)(kp);      b0.q[1] = *(const uint4*)(kp + 16);
      b1.q[0] = *(const uint4*)(kp + 32); b1.q[1] = *(const uint4*)(kp + 48);
      s0 = wmma_bf16(qa0.v, b0.v, s0);
      s0 = wmma_bf16(qa1.v, b1.v, s0);
    }
    {
      const unsigned short* kp = kbase + (size_t)(j + 16 + ln16) * HD_ + hi * 8;
      Frag b0, b1;
      b0.q[0] = *(const uint4*)(kp);      b0.q[1] = *(const uint4*)(kp + 16);
      b1.q[0] = *(const uint4*)(kp + 32); b1.q[1] = *(const uint4*)(kp + 48);
      s1 = wmma_bf16(qa0.v, b0.v, s1);
      s1 = wmma_bf16(qa1.v, b1.v, s1);
    }
    // fixed-shift exp; accumulate this lane's partial row sums; stage P in LDS
#pragma unroll
    for (int v = 0; v < 8; ++v) {
      const float p0 = __expf(s0[v] - shift);
      const float p1 = __expf(s1[v] - shift);
      lst[v] += p0 + p1;
      const int r = v + 8 * hi;                 // row this half owns in register v
      prow[r * 32 + ln16]      = f2bf(p0);
      prow[r * 32 + 16 + ln16] = f2bf(p1);
    }
    // re-read P as an A fragment (16x32) from LDS
    Frag pa;
    const unsigned short* pp = prow + ln16 * 32 + hi * 8;
    pa.q[0] = *(const uint4*)(pp);
    pa.q[1] = *(const uint4*)(pp + 16);
#pragma unroll
    for (int nn = 0; nn < 4; ++nn) {
      const unsigned short* vp = vbase + (size_t)(nn * 16 + ln16) * L_ + j + hi * 8;
      Frag vb;
      vb.q[0] = *(const uint4*)(vp);
      vb.q[1] = *(const uint4*)(vp + 16);
      oacc[nn] = wmma_bf16(pa.v, vb.v, oacc[nn]);
    }
  }
  const float hs = head_scale[h];
#pragma unroll
  for (int v = 0; v < 8; ++v) {
    float rs = lst[v];                           // full row sum: reduce 16 lanes once
#pragma unroll
    for (int off = 1; off < 16; off <<= 1) rs += __shfl_xor(rs, off);
    const float inv = hs / rs;
    const int l = qtile * 16 + v + 8 * hi;
#pragma unroll
    for (int nn = 0; nn < 4; ++nn) {
      const int c = h * HD_ + nn * 16 + ln16;
      O[((size_t)(b * L_ + l)) * C_ + c] = f2bf(oacc[nn][v] * inv);
    }
  }
}

// ---------------------------------------------------------------- launch
extern "C" void kernel_launch(void* const* d_in, const int* in_sizes, int n_in,
                              void* d_out, int out_size, void* d_ws, size_t ws_size,
                              hipStream_t stream) {
  const float* x       = (const float*)d_in[0];
  const float* in_w    = (const float*)d_in[1];
  const float* in_b    = (const float*)d_in[2];
  const float* logit_s = (const float*)d_in[3];
  const float* head_s  = (const float*)d_in[4];
  const float* out_w   = (const float*)d_in[5];
  const float* out_b   = (const float*)d_in[6];
  const float* ln1_g   = (const float*)d_in[7];
  const float* ln1_b   = (const float*)d_in[8];
  const float* lna_g   = (const float*)d_in[9];
  const float* lna_b   = (const float*)d_in[10];
  const float* ls1     = (const float*)d_in[11];
  const float* ln2_g   = (const float*)d_in[12];
  const float* ln2_b   = (const float*)d_in[13];
  const float* fc_w    = (const float*)d_in[14];
  const float* fc_b    = (const float*)d_in[15];
  const float* proj_w  = (const float*)d_in[16];
  const float* proj_b  = (const float*)d_in[17];
  const float* ls2     = (const float*)d_in[18];
  float* out = (float*)d_out;

  // workspace partition (~185 MB)
  char* p = (char*)d_ws;
  auto alloc = [&](size_t bytes) -> char* {
    char* r = p;
    p += (bytes + 255) & ~(size_t)255;
    return r;
  };
  const int BL = B_ * L_;  // 4096
  unsigned short* wqkv_bf = (unsigned short*)alloc((size_t)3 * C_ * C_ * 2);
  unsigned short* wout_bf = (unsigned short*)alloc((size_t)C_ * C_ * 2);
  unsigned short* wfc_bf  = (unsigned short*)alloc((size_t)MLP_ * C_ * 2);
  unsigned short* wpr_bf  = (unsigned short*)alloc((size_t)C_ * MLP_ * 2);
  unsigned short* xln_bf  = (unsigned short*)alloc((size_t)BL * C_ * 2);
  float*          qkv     = (float*)alloc((size_t)BL * 3 * C_ * 4);
  unsigned short* qn      = (unsigned short*)alloc((size_t)BL * C_ * 2);
  unsigned short* kn      = (unsigned short*)alloc((size_t)BL * C_ * 2);
  unsigned short* vT      = (unsigned short*)alloc((size_t)BL * C_ * 2);
  unsigned short* obf     = (unsigned short*)alloc((size_t)BL * C_ * 2);
  float*          attn    = (float*)alloc((size_t)BL * C_ * 4);
  float*          x2      = (float*)alloc((size_t)BL * C_ * 4);
  unsigned short* h2bf    = (unsigned short*)alloc((size_t)BL * C_ * 2);
  unsigned short* hmid    = (unsigned short*)alloc((size_t)BL * MLP_ * 2);

  // weights -> bf16
  k_cvt_bf16<<<(3 * C_ * C_ + 255) / 256, 256, 0, stream>>>(in_w, wqkv_bf, 3 * C_ * C_);
  k_cvt_bf16<<<(C_ * C_ + 255) / 256, 256, 0, stream>>>(out_w, wout_bf, C_ * C_);
  k_cvt_bf16<<<(MLP_ * C_ + 255) / 256, 256, 0, stream>>>(fc_w, wfc_bf, MLP_ * C_);
  k_cvt_bf16<<<(C_ * MLP_ + 255) / 256, 256, 0, stream>>>(proj_w, wpr_bf, C_ * MLP_);
  // LN1 -> bf16 activations
  k_ln_bf16<<<BL, 256, 0, stream>>>(x, ln1_g, ln1_b, xln_bf);
  // qkv = ln1(x) @ in_proj_w^T + in_proj_b   [4096 x 3072]
  k_gemm<0><<<dim3(3 * C_ / 64, BL / 128), 256, 0, stream>>>(
      xln_bf, wqkv_bf, in_b, BL, 3 * C_, C_, qkv, nullptr, nullptr, nullptr);
  // split heads, l2-normalize q/k (fold logit scale), transpose v
  k_split_qkv<<<(B_ * H_ * L_) / 8, 256, 0, stream>>>(qkv, logit_s, qn, kn, vT);
  // scaled-cosine flash attention (head_scale folded in)
  k_flash<<<dim3(L_ / 16 / 8, B_ * H_), 256, 0, stream>>>(qn, kn, vT, logit_s, head_s, obf);
  // attn out-proj: attn = O @ out_w^T + out_b
  k_gemm<0><<<dim3(C_ / 64, BL / 128), 256, 0, stream>>>(
      obf, wout_bf, out_b, BL, C_, C_, attn, nullptr, nullptr, nullptr);
  // x2 = x + ls1 * LN_attn(attn)
  k_resid_ln<<<BL, 256, 0, stream>>>(attn, x, lna_g, lna_b, ls1, x2);
  // h = LN2(x2) -> bf16
  k_ln_bf16<<<BL, 256, 0, stream>>>(x2, ln2_g, ln2_b, h2bf);
  // hmid = GELU(h @ fc_w^T + fc_b) -> bf16   [4096 x 4096]
  k_gemm<1><<<dim3(MLP_ / 64, BL / 128), 256, 0, stream>>>(
      h2bf, wfc_bf, fc_b, BL, MLP_, C_, nullptr, hmid, nullptr, nullptr);
  // out = x2 + ls2 * (hmid @ proj_w^T + proj_b)
  k_gemm<2><<<dim3(C_ / 64, BL / 128), 256, 0, stream>>>(
      hmid, wpr_bf, proj_b, BL, C_, MLP_, out, nullptr, x2, ls2);

  (void)in_sizes; (void)n_in; (void)out_size; (void)ws_size;
}
